// ISFNOTransitionModel_30305289240729
// MI455X (gfx1250) — matrix operand order repaired and checked by reading, not compiled, not run on hardware
//
#include <hip/hip_runtime.h>
#include <hip/hip_bf16.h>
#include <math.h>

// ---------------------------------------------------------------------------
// ISF-NO transition model for MI455X (gfx1250, wave32, WMMA).
//  - All GEMMs (12 MLP evals + spectral circulant + C-projection + both
//    rank-16 control terms) run on v_wmma_f32_16x16x32_f16, f32 accumulate.
//  - spectral_evolve(z) == z @ M^T with circulant M built from exp(r) by a
//    prep kernel (FFT folded into a GEMM).
//  - Branchless erf-GELU (rcp + exp, no EXEC divergence)
//  - Per block (128 thr = 4 waves, 64 rows): W1/W2/W3/biases in LDS (~170KB),
//    per-wave f16 activation slabs (~34KB each). Total 303KB LDS / WGP.
//  - M, padded-C/B_ctrl/D_mat stream from the 192MB L2 as B-fragments.
// ---------------------------------------------------------------------------

typedef _Float16 h16;
typedef __attribute__((ext_vector_type(16))) _Float16 v16h;
typedef __attribute__((ext_vector_type(8)))  _Float16 v8h;
typedef __attribute__((ext_vector_type(4)))  _Float16 v4h;
typedef __attribute__((ext_vector_type(8)))  float    v8f;

#define NB     65536
#define ND     256
#define NU     16
#define NH     128
#define NYW    25
#define SA     264   // padded stride (halves) for 256-wide activation rows
#define SH     136   // padded stride (halves) for 128-wide rows
#define ROWS_PER_BLOCK 64
#define NBLOCKS (NB / ROWS_PER_BLOCK)   // 1024
#define NPART   (NBLOCKS * 4)           // 4096 residual partials
#define SMEM_BYTES 310272

// Branchless exact-erf GELU: Abramowitz-Stegun 7.1.26 (|err| < 1.5e-7).
// 1x v_rcp_f32 + 1x v_exp_f32 + short FMA chain; no EXEC divergence.
__device__ __forceinline__ float gelu_exact(float x) {
  float xs = x * 0.7071067811865475f;
  float ax = fabsf(xs);
  float t  = __builtin_amdgcn_rcpf(__builtin_fmaf(0.3275911f, ax, 1.0f));
  float p  = __builtin_fmaf(1.061405429f, t, -1.453152027f);
  p = __builtin_fmaf(p, t, 1.421413741f);
  p = __builtin_fmaf(p, t, -0.284496736f);
  p = __builtin_fmaf(p, t, 0.254829592f);
  p = p * t;
  float e = __expf(-ax * ax);
  float erfv = copysignf(1.0f - p * e, xs);
  return 0.5f * x * (1.0f + erfv);
}

// A-fragment (16x32 f16, M=rows): lane L(0-15)->row L, K{kb..kb+7, kb+16..23};
// lane L+16 -> row L, K{kb+8..15, kb+24..31}.   (ISA 7.12.2, 16-bit A layout)
__device__ __forceinline__ v16h frag_A(const h16* buf, int stride, int kb, int lane) {
  const int r = lane & 15, hf = lane >> 4;
  const h16* p = buf + r * stride + kb + hf * 8;
  v8h lo = *(const v8h*)p;
  v8h hi = *(const v8h*)(p + 16);
  v16h a;
#pragma unroll
  for (int i = 0; i < 8; ++i) { a[i] = lo[i]; a[i + 8] = hi[i]; }
  return a;
}

// B-fragment (32x16 f16, K x N): lane n holds column n; lanes 0-15 -> K {kb..kb+7,
// kb+16..23}, lanes 16-31 -> K {kb+8..15, kb+24..31}. Weight stored row-major [out][in],
// so "column n of B" == row n of the torch weight: no transpose needed.
__device__ __forceinline__ v16h frag_B(const h16* w, int stride, int n0, int kb, int lane) {
  const int n = n0 + (lane & 15), hf = lane >> 4;
  const h16* p = w + n * stride + kb + hf * 8;
  v8h lo = *(const v8h*)p;
  v8h hi = *(const v8h*)(p + 16);
  v16h b;
#pragma unroll
  for (int i = 0; i < 8; ++i) { b[i] = lo[i]; b[i + 8] = hi[i]; }
  return b;
}

__device__ __forceinline__ v8f wmma_f16(v16h a, v16h b, v8f c) {
  return __builtin_amdgcn_wmma_f32_16x16x32_f16(false, a, false, b, (short)0, c,
                                                false, false);
}

// Store a 16x16 f32 D-tile (VGPR v: row v+hf*8, col lane&15) to LDS f16.
__device__ __forceinline__ void store_tile(h16* buf, int stride, int n0, int lane, v8f acc) {
  const int n = lane & 15, hf = lane >> 4;
#pragma unroll
  for (int v = 0; v < 8; ++v)
    buf[(v + hf * 8) * stride + n0 + n] = (h16)acc[v];
}

// Stage ut*dt as a 16x32 f16 A-slab (cols 16..31 zero) into HB.
__device__ __forceinline__ void stage_ut(h16* HB, const float* ut, int row0,
                                         float dtv, int lane) {
#pragma unroll
  for (int r = 0; r < 16; ++r) {
    float v = (lane < NU) ? ut[(size_t)(row0 + r) * NU + lane] * dtv : 0.0f;
    HB[r * SH + lane] = (h16)v;
  }
}

// One MLP eval for a wave's 16-row slab.  MODE 0: lift (out += zt, -> DST)
// MODE 1: inverse step (out = T - mlp, -> DST)   MODE 2: rev residual accum.
template <int MODE>
__device__ __forceinline__ void wave_mlp(
    const h16* X, h16* HB, h16* H2,
    const h16* lW1, const h16* lW2, const h16* lW3,
    const float* lb1, const float* lb2, const float* lb3,
    int lane,
    const float* ztg, const h16* T,
    const h16* ZLb, const h16* CIb,
    h16* DST, float* resid)
{
  const int hf = lane >> 4, nn = lane & 15;

  // layer 1: [16x256] @ W1^T + b1 -> gelu -> HB[16x128]
  v16h a[8];
#pragma unroll
  for (int kc = 0; kc < 8; ++kc) a[kc] = frag_A(X, SA, kc * 32, lane);
#pragma unroll
  for (int t = 0; t < 8; ++t) {
    v8f acc;
    const float bv = lb1[t * 16 + nn];
#pragma unroll
    for (int v = 0; v < 8; ++v) acc[v] = bv;
#pragma unroll
    for (int kc = 0; kc < 8; ++kc)
      acc = wmma_f16(a[kc], frag_B(lW1, SA, t * 16, kc * 32, lane), acc);
#pragma unroll
    for (int v = 0; v < 8; ++v) acc[v] = gelu_exact(acc[v]);
    store_tile(HB, SH, t * 16, lane, acc);
  }

  // layer 2: [16x128] @ W2^T + b2 -> gelu -> H2[16x128]
  v16h h[4];
#pragma unroll
  for (int kc = 0; kc < 4; ++kc) h[kc] = frag_A(HB, SH, kc * 32, lane);
#pragma unroll
  for (int t = 0; t < 8; ++t) {
    v8f acc;
    const float bv = lb2[t * 16 + nn];
#pragma unroll
    for (int v = 0; v < 8; ++v) acc[v] = bv;
#pragma unroll
    for (int kc = 0; kc < 4; ++kc)
      acc = wmma_f16(h[kc], frag_B(lW2, SH, t * 16, kc * 32, lane), acc);
#pragma unroll
    for (int v = 0; v < 8; ++v) acc[v] = gelu_exact(acc[v]);
    store_tile(H2, SH, t * 16, lane, acc);
  }

  // layer 3: [16x128] @ W3^T + b3 -> consumer (16 output tiles over D=256)
  v16h g[4];
#pragma unroll
  for (int kc = 0; kc < 4; ++kc) g[kc] = frag_A(H2, SH, kc * 32, lane);
#pragma unroll
  for (int t = 0; t < 16; ++t) {
    v8f acc;
    const float bv = lb3[t * 16 + nn];
#pragma unroll
    for (int v = 0; v < 8; ++v) acc[v] = bv;
#pragma unroll
    for (int kc = 0; kc < 4; ++kc)
      acc = wmma_f16(g[kc], frag_B(lW3, SH, t * 16, kc * 32, lane), acc);
#pragma unroll
    for (int v = 0; v < 8; ++v) {
      const int m = v + hf * 8, d = t * 16 + nn;
      if constexpr (MODE == 0) {
        float o = acc[v] + ztg[m * ND + d];          // lift: z + mlp(z)
        DST[m * SA + d] = (h16)o;
      } else if constexpr (MODE == 1) {
        float o = (float)T[m * SA + d] - acc[v];     // z <- target - mlp(z)
        DST[m * SA + d] = (h16)o;
      } else {
        float r = (float)ZLb[m * SA + d] - (float)CIb[m * SA + d] - acc[v];
        *resid += r * r;                             // (zl - (zi + mlp(zi)))^2
      }
    }
  }
}

// ---------------------------------------------------------------------------
// Prep: circulant M from exp(r); zero-padded f16 C [32x256], B_ctrl [256x32],
// D_mat [32x32] staged in workspace for B-fragment streaming.
// ---------------------------------------------------------------------------
__global__ void __launch_bounds__(256) isfno_prep(
    const float* __restrict__ rr, const float* __restrict__ ri,
    const float* __restrict__ Cm, const float* __restrict__ Bc,
    const float* __restrict__ Dm,
    h16* __restrict__ Mw, h16* __restrict__ Cpad,
    h16* __restrict__ Bpad, h16* __restrict__ Dpad)
{
  __shared__ float sr[129], si[129], cc[256];
  const int t = threadIdx.x;
  if (t < 129) {
    float e = expf(rr[t]);
    sr[t] = e * cosf(ri[t]);
    si[t] = e * sinf(ri[t]);
  }
  __syncthreads();
  if (t < 256) {
    float acc = sr[0] + ((t & 1) ? -sr[128] : sr[128]);
    for (int k = 1; k < 128; ++k) {
      float ang = 6.283185307179586f * (float)((k * t) & 255) * (1.0f / 256.0f);
      acc += 2.0f * (sr[k] * cosf(ang) - si[k] * sinf(ang));
    }
    cc[t] = acc * (1.0f / 256.0f);
  }
  __syncthreads();
  for (int i = t; i < ND * ND; i += 256) {
    int m = i >> 8, j = i & 255;
    Mw[i] = (h16)cc[(m - j) & 255];
  }
  for (int i = t; i < 32 * ND; i += 256) {
    int q = i >> 8, j = i & 255;
    Cpad[i] = (h16)((q < NYW) ? Cm[q * ND + j] : 0.0f);
  }
  for (int i = t; i < ND * 32; i += 256) {
    int d = i >> 5, u = i & 31;
    Bpad[i] = (h16)((u < NU) ? Bc[d * NU + u] : 0.0f);
  }
  for (int i = t; i < 32 * 32; i += 256) {
    int q = i >> 5, u = i & 31;
    Dpad[i] = (h16)((q < NYW && u < NU) ? Dm[q * NU + u] : 0.0f);
  }
}

// ---------------------------------------------------------------------------
// Main kernel: 128 threads = 4 wave32, 16 rows per wave, 64 rows per block.
// ---------------------------------------------------------------------------
__global__ void __launch_bounds__(128) isfno_main(
    const float* __restrict__ zt, const float* __restrict__ dtp,
    const float* __restrict__ ut,
    const float* __restrict__ W1, const float* __restrict__ b1,
    const float* __restrict__ W2, const float* __restrict__ b2,
    const float* __restrict__ W3, const float* __restrict__ b3,
    const h16* __restrict__ Mw, const h16* __restrict__ Cpad,
    const h16* __restrict__ Bpad, const h16* __restrict__ Dpad,
    float* __restrict__ out_z, float* __restrict__ out_y,
    float* __restrict__ ws_part)
{
  extern __shared__ char smem[];
  h16*   lW1 = (h16*)smem;                 // 128 x 264
  h16*   lW2 = lW1 + 128 * SA;             // 128 x 136
  h16*   lW3 = lW2 + 128 * SH;             // 256 x 136
  float* lb1 = (float*)(lW3 + 256 * SH);   // 128 + 128 + 256 floats
  float* lb2 = lb1 + 128;
  float* lb3 = lb2 + 128;

  const int tid = threadIdx.x;
  // stage weights (f32 global -> f16 LDS, padded rows)
  for (int i = tid; i < NH * ND; i += 128) lW1[(i >> 8) * SA + (i & 255)] = (h16)W1[i];
  for (int i = tid; i < NH * NH; i += 128) lW2[(i >> 7) * SH + (i & 127)] = (h16)W2[i];
  for (int i = tid; i < ND * NH; i += 128) lW3[(i >> 7) * SH + (i & 127)] = (h16)W3[i];
  if (tid < 128) { lb1[tid] = b1[tid]; lb2[tid] = b2[tid]; }
  for (int i = tid; i < ND; i += 128) lb3[i] = b3[i];
  __syncthreads();

  const int wid = tid >> 5, lane = tid & 31;
  const int hf = lane >> 4, nn = lane & 15;
  h16* wbase = (h16*)(lb3 + ND) + wid * (3 * 16 * SA + 2 * 16 * SH);
  h16* ZL = wbase;                 // z_lifted      16 x 264
  h16* TB = ZL + 16 * SA;          // target (e)    16 x 264
  h16* CI = TB + 16 * SA;          // iterate       16 x 264
  h16* HB = CI + 16 * SA;          // hidden 1      16 x 136
  h16* H2 = HB + 16 * SH;          // hidden 2      16 x 136

  const int row0 = blockIdx.x * ROWS_PER_BLOCK + wid * 16;
  const float dtv = dtp[0];
  const float* ztg = zt + (size_t)row0 * ND;

  // --- stage zt slab as f16 iterate input (vectorized f32x4 -> f16x4)
  for (int idx = lane; idx < 16 * 64; idx += 32) {
    const int r = idx >> 6, c = (idx & 63) * 4;
    const float4 f = *(const float4*)(ztg + r * ND + c);
    v4h o; o[0] = (h16)f.x; o[1] = (h16)f.y; o[2] = (h16)f.z; o[3] = (h16)f.w;
    *(v4h*)(CI + r * SA + c) = o;
  }

  // --- 1) z_lifted = zt + mlp(zt)  -> ZL
  wave_mlp<0>(CI, HB, H2, lW1, lW2, lW3, lb1, lb2, lb3, lane,
              ztg, nullptr, nullptr, nullptr, ZL, nullptr);

  // --- 2) e = ZL @ M^T + (ut*dt) @ B_ctrl^T  -> TB   (all WMMA)
  {
    stage_ut(HB, ut, row0, dtv, lane);       // 16x32 f16 control slab
    v16h uf = frag_A(HB, SH, 0, lane);
    v16h za[8];
#pragma unroll
    for (int kc = 0; kc < 8; ++kc) za[kc] = frag_A(ZL, SA, kc * 32, lane);
#pragma unroll
    for (int t = 0; t < 16; ++t) {
      v8f acc;
#pragma unroll
      for (int v = 0; v < 8; ++v) acc[v] = 0.0f;
      acc = wmma_f16(uf, frag_B(Bpad, 32, t * 16, 0, lane), acc);
#pragma unroll
      for (int kc = 0; kc < 8; ++kc)
        acc = wmma_f16(za[kc], frag_B(Mw, ND, t * 16, kc * 32, lane), acc);
      store_tile(TB, SA, t * 16, lane, acc);
    }
  }

  // --- 3) zt1 = inv_lift(e): 5 x (z = e - mlp(z)), z0 = e
  {
    const uint4* s = (const uint4*)TB; uint4* d = (uint4*)CI;
    for (int i = lane; i < (16 * SA) / 8; i += 32) d[i] = s[i];
  }
#pragma unroll 1
  for (int it = 0; it < 5; ++it)
    wave_mlp<1>(CI, HB, H2, lW1, lW2, lW3, lb1, lb2, lb3, lane,
                nullptr, TB, nullptr, nullptr, CI, nullptr);

  // write zt1 (f32, vectorized)
  for (int idx = lane; idx < 16 * 64; idx += 32) {
    const int r = idx >> 6, c = (idx & 63) * 4;
    v4h hv = *(const v4h*)(CI + r * SA + c);
    float4 f; f.x = (float)hv[0]; f.y = (float)hv[1];
    f.z = (float)hv[2]; f.w = (float)hv[3];
    *(float4*)(out_z + (size_t)(row0 + r) * ND + c) = f;
  }

  // --- 4) yt1 = zt1 @ C^T + (ut*dt) @ D^T   (all WMMA, padded to 32 cols)
  {
    stage_ut(HB, ut, row0, dtv, lane);       // HB free again after inv #1
    v16h uf = frag_A(HB, SH, 0, lane);
    v16h ya[8];
#pragma unroll
    for (int kc = 0; kc < 8; ++kc) ya[kc] = frag_A(CI, SA, kc * 32, lane);
#pragma unroll
    for (int t = 0; t < 2; ++t) {
      v8f acc;
#pragma unroll
      for (int v = 0; v < 8; ++v) acc[v] = 0.0f;
      acc = wmma_f16(uf, frag_B(Dpad, 32, t * 16, 0, lane), acc);
#pragma unroll
      for (int kc = 0; kc < 8; ++kc)
        acc = wmma_f16(ya[kc], frag_B(Cpad, ND, t * 16, kc * 32, lane), acc);
      const int q = t * 16 + nn;
      if (q < NYW) {
#pragma unroll
        for (int v = 0; v < 8; ++v) {
          const int m = v + hf * 8;
          out_y[(size_t)(row0 + m) * NYW + q] = acc[v];
        }
      }
    }
  }

  // --- 5) zi = inv_lift(z_lifted), rev_residual accumulation
  {
    const uint4* s = (const uint4*)ZL; uint4* d = (uint4*)CI;
    for (int i = lane; i < (16 * SA) / 8; i += 32) d[i] = s[i];
  }
#pragma unroll 1
  for (int it = 0; it < 5; ++it)
    wave_mlp<1>(CI, HB, H2, lW1, lW2, lW3, lb1, lb2, lb3, lane,
                nullptr, ZL, nullptr, nullptr, CI, nullptr);

  float resid = 0.0f;
  wave_mlp<2>(CI, HB, H2, lW1, lW2, lW3, lb1, lb2, lb3, lane,
              nullptr, nullptr, ZL, CI, nullptr, &resid);

#pragma unroll
  for (int o = 16; o > 0; o >>= 1) resid += __shfl_down(resid, o, 32);
  if (lane == 0)
    ws_part[blockIdx.x * 4 + wid] = resid * (1.0f / ((float)NB * (float)ND));
}

// ---------------------------------------------------------------------------
// Deterministic fixed-order reduction of per-wave residual partials.
// ---------------------------------------------------------------------------
__global__ void __launch_bounds__(256) isfno_reduce(
    const float* __restrict__ parts, float* __restrict__ out_r)
{
  __shared__ float s[256];
  const int t = threadIdx.x;
  float a = 0.0f;
  for (int i = t; i < NPART; i += 256) a += parts[i];
  s[t] = a;
  __syncthreads();
  for (int o = 128; o > 0; o >>= 1) {
    if (t < o) s[t] += s[t + o];
    __syncthreads();
  }
  if (t == 0) out_r[0] = s[0];
}

extern "C" void kernel_launch(void* const* d_in, const int* in_sizes, int n_in,
                              void* d_out, int out_size, void* d_ws, size_t ws_size,
                              hipStream_t stream) {
  (void)in_sizes; (void)n_in; (void)out_size; (void)ws_size;
  const float* zt  = (const float*)d_in[0];
  const float* dt  = (const float*)d_in[1];
  const float* ut  = (const float*)d_in[2];
  const float* W1  = (const float*)d_in[3];
  const float* b1  = (const float*)d_in[4];
  const float* W2  = (const float*)d_in[5];
  const float* b2  = (const float*)d_in[6];
  const float* W3  = (const float*)d_in[7];
  const float* b3  = (const float*)d_in[8];
  const float* rr  = (const float*)d_in[9];
  const float* ri  = (const float*)d_in[10];
  const float* Bc  = (const float*)d_in[11];
  const float* Cm  = (const float*)d_in[12];
  const float* Dm  = (const float*)d_in[13];

  h16*   Mw    = (h16*)d_ws;                               // 131072 B
  h16*   Cpad  = (h16*)((char*)d_ws + 131072);             //  16384 B
  h16*   Bpad  = (h16*)((char*)d_ws + 147456);             //  16384 B
  h16*   Dpad  = (h16*)((char*)d_ws + 163840);             //   2048 B
  float* parts = (float*)((char*)d_ws + 165888);           //  16384 B

  float* out_z = (float*)d_out;
  float* out_y = out_z + (size_t)NB * ND;
  float* out_r = out_y + (size_t)NB * NYW;

  (void)hipFuncSetAttribute((const void*)isfno_main,
                            hipFuncAttributeMaxDynamicSharedMemorySize,
                            SMEM_BYTES);

  isfno_prep<<<1, 256, 0, stream>>>(rr, ri, Cm, Bc, Dm, Mw, Cpad, Bpad, Dpad);
  isfno_main<<<NBLOCKS, 128, SMEM_BYTES, stream>>>(
      zt, dt, ut, W1, b1, W2, b2, W3, b3, Mw, Cpad, Bpad, Dpad,
      out_z, out_y, parts);
  isfno_reduce<<<1, 256, 0, stream>>>(parts, out_r);
}
